// ImprovedGateNetwork_23957327577720
// MI455X (gfx1250) — matrix operand order repaired and checked by reading, not compiled, not run on hardware
//
#include <hip/hip_runtime.h>
#include <hip/hip_bf16.h>

// ---------------------------------------------------------------------------
// Problem constants (from reference): V=40, D=128, H=64, T=40, B=8192, E=4
// FEAT = 2H + 1 + 4 + 6 + 4 = 143  (padded to 160 = 5 K-chunks of 32 for WMMA)
// ---------------------------------------------------------------------------
#define BB   8192
#define TT   40
#define VV   40
#define DD   128
#define HH   64
#define EE   4
#define FPAD 160

typedef __attribute__((ext_vector_type(16))) _Float16 v16h;
typedef __attribute__((ext_vector_type(8)))  float    v8f;

union Frag {                // one WMMA A/B operand (16 f16 per lane)
  v16h  v;
  _Float16 h[16];
  unsigned u[8];
  uint4 q[2];
};

// ws layout:
//   [0, 128KB)   : LSTM B-fragments [dir(2)][ntile(16)][kchunk(4)][lane(32)] x 32B
//   [128KB, ...) : featf16  f16 [B][160]  (cols 0..127 = h_f|h_b, 128..142 stats, 143..159 = 0)
#define BFRAG_BYTES (2 * 16 * 4 * 32 * 32)

__device__ __forceinline__ float frcp(float x) { return __builtin_amdgcn_rcpf(x); }
// branchless fast sigmoid: one v_exp + one v_rcp
__device__ __forceinline__ float fsig(float x) {
  x = fminf(fmaxf(x, -30.f), 30.f);
  return frcp(1.f + __expf(-x));
}
__device__ __forceinline__ float ftanh(float x) { return 2.f * fsig(2.f * x) - 1.f; }

// A-fragment (16-bit 16x32, wave32) K offset within a K=32 chunk (contiguous, matches B):
//   lane L: row M = L%16; VGPR v -> K = 2v; +16 for lanes 16..31
__device__ __forceinline__ int a_kbase(int v, int halfId) {
  return 2 * v + halfId * 16;
}

__device__ __forceinline__ v8f wmma_f16(const Frag& a, const Frag& b, v8f c) {
  return __builtin_amdgcn_wmma_f32_16x16x32_f16(false, a.v, false, b.v, (short)0, c,
                                                false, false);
}

// ---------------------------------------------------------------------------
// Kernel 1: pack LSTM B-side fragments.
//   Concatenated B (K=128, N=256 per direction):
//     K in [0,40)   : P'[K][N] = sum_d emb[K][d]*W_ih[N][d] + bias[N]  (bias baked in:
//                     one-hot has exactly one 1 per row, pad row 0 -> bias, as reference)
//     K in [40,64)  : 0
//     K in [64,128) : Whh^T[K-64][N] = W_hh[N][K-64]
//   B-fragment layout (16-bit 32x16, wave32): lane L holds col N = L%16,
//   halves j=0..15 hold K = (L>>4)*16 + j within the chunk.
// ---------------------------------------------------------------------------
__global__ void setup_bfrag(const float* __restrict__ emb,
                            const float* __restrict__ Wihf, const float* __restrict__ Whhf,
                            const float* __restrict__ bf,
                            const float* __restrict__ Wihb, const float* __restrict__ Whhb,
                            const float* __restrict__ bb,
                            unsigned char* __restrict__ bfrag) {
  int gid = blockIdx.x * blockDim.x + threadIdx.x;   // 4096 threads total
  if (gid >= 4096) return;
  int lane = gid & 31;
  int kc   = (gid >> 5) & 3;
  int nt   = (gid >> 7) & 15;
  int dir  = (gid >> 11) & 1;
  const float* Wih  = dir ? Wihb : Wihf;
  const float* Whh  = dir ? Whhb : Whhf;
  const float* bias = dir ? bb   : bf;
  int N = nt * 16 + (lane & 15);
  Frag f;
  for (int j = 0; j < 16; ++j) {
    int K = kc * 32 + ((lane >> 4) << 4) + j;
    float val;
    if (K < 64) {
      if (K < VV) {
        float s = bias[N];
        for (int d = 0; d < DD; ++d) s += emb[K * DD + d] * Wih[N * DD + d];
        val = s;
      } else {
        val = 0.f;
      }
    } else {
      val = Whh[N * HH + (K - 64)];
    }
    f.h[j] = (_Float16)val;
  }
  uint4* p = (uint4*)(bfrag + gid * 32);
  p[0] = f.q[0];
  p[1] = f.q[1];
}

// ---------------------------------------------------------------------------
// Kernel 2: bidirectional LSTM recurrence via WMMA.
//   Grid: B/16 = 512 workgroups of 128 threads (4 waves), one 16-row batch tile
//   per workgroup. Wave w owns gate quarter w (torch order i,f,g,o) = N-tiles
//   4w..4w+3; its 16 B-fragments are VGPR-resident for the whole scan.
//   Per step per wave: 16 v_wmma_f32_16x16x32_f16 over concat K=128 =
//   [onehot(x)64 | h64]. Activations are branchless (tanh = 2*sigma(2x)-1 with
//   wave-uniform scale/offset) + fast v_rcp.
//   Final h (f16) is written straight into featf16 cols [dir*64, dir*64+64).
// ---------------------------------------------------------------------------
__global__ __launch_bounds__(128) void lstm_kernel(const int* __restrict__ x,
                                                   const unsigned char* __restrict__ bfrag,
                                                   _Float16* __restrict__ featf16) {
  __shared__ _Float16 h16[16][64];        // h state, f16, K-contiguous
  __shared__ float    q4[4][16][64];      // activated gate quarters
  __shared__ int      xs[16][TT];

  const int tid    = threadIdx.x;
  const int lane   = tid & 31;
  const int w      = tid >> 5;
  const int rowb   = blockIdx.x * 16;
  const int m      = lane & 15;
  const int halfId = lane >> 4;

  // wave-uniform activation constants: quarter 2 (g) is tanh, others sigmoid
  const float asc  = (w == 2) ? 2.f : 1.f;
  const float aoff = (w == 2) ? -1.f : 0.f;

  for (int i = tid; i < 16 * TT; i += 128) {
    int r = i / TT, t = i % TT;
    xs[r][t] = x[(rowb + r) * TT + t];
  }

  for (int dir = 0; dir < 2; ++dir) {
    Frag Bf[4][4];
#pragma unroll
    for (int n = 0; n < 4; ++n)
#pragma unroll
      for (int k = 0; k < 4; ++k) {
        const uint4* p =
            (const uint4*)(bfrag + (size_t)((((dir * 16 + (w * 4 + n)) * 4 + k) * 32 + lane)) * 32);
        Bf[n][k].q[0] = p[0];
        Bf[n][k].q[1] = p[1];
      }
    for (int i = tid; i < 16 * 64; i += 128) ((_Float16*)h16)[i] = (_Float16)0.f;
    float c[8], hreg[8];
#pragma unroll
    for (int e = 0; e < 8; ++e) { c[e] = 0.f; hreg[e] = 0.f; }
    __syncthreads();

    for (int s = 0; s < TT; ++s) {
      const int t  = dir ? (TT - 1 - s) : s;
      const int xv = xs[m][t];

      Frag A[4];
#pragma unroll
      for (int cc = 0; cc < 2; ++cc)
#pragma unroll
        for (int j = 0; j < 16; ++j) {
          int K = cc * 32 + a_kbase(j >> 1, halfId) + (j & 1);
          A[cc].h[j] = (K == xv) ? (_Float16)1.f : (_Float16)0.f;
        }
#pragma unroll
      for (int cc = 0; cc < 2; ++cc)
#pragma unroll
        for (int v = 0; v < 8; ++v) {
          int col = cc * 32 + a_kbase(v, halfId);
          A[2 + cc].u[v] = *(const unsigned*)&h16[m][col];
        }

      v8f acc[4];
#pragma unroll
      for (int n = 0; n < 4; ++n) {
        v8f a = {};
#pragma unroll
        for (int k = 0; k < 4; ++k) a = wmma_f16(A[k], Bf[n][k], a);
        acc[n] = a;
      }

#pragma unroll
      for (int n = 0; n < 4; ++n)
#pragma unroll
        for (int r = 0; r < 8; ++r) {
          float av = asc * fsig(asc * acc[n][r]) + aoff;   // branchless sigmoid/tanh
          q4[w][r + halfId * 8][n * 16 + m] = av;
        }
      __syncthreads();

#pragma unroll
      for (int e = 0; e < 8; ++e) {
        int idx = lane + 32 * e;
        int row = 4 * w + (idx >> 6);
        int col = idx & 63;
        float iv = q4[0][row][col];
        float fv = q4[1][row][col];
        float gv = q4[2][row][col];
        float ov = q4[3][row][col];
        c[e]     = fv * c[e] + iv * gv;
        float hv = ov * ftanh(c[e]);
        hreg[e]  = hv;
        h16[row][col] = (_Float16)hv;
      }
      __syncthreads();
    }

#pragma unroll
    for (int e = 0; e < 8; ++e) {
      int idx = lane + 32 * e;
      int row = 4 * w + (idx >> 6);
      int col = idx & 63;
      featf16[(size_t)(rowb + row) * FPAD + dir * 64 + col] = (_Float16)hreg[e];
    }
    __syncthreads();
  }
}

// ---------------------------------------------------------------------------
// Kernel 3: statistical features, one thread per row -> featf16 cols 128..159.
// ---------------------------------------------------------------------------
__global__ __launch_bounds__(256) void stats_kernel(const int* __restrict__ x,
                                                    _Float16* __restrict__ featf16) {
  const int b = blockIdx.x * 256 + threadIdx.x;
  if (b >= BB) return;

  int xr[TT];
#pragma unroll
  for (int t = 0; t < TT; ++t) xr[t] = x[b * TT + t];

  float Lf = 0.f;
#pragma unroll
  for (int t = 0; t < TT; ++t) Lf += (xr[t] != 0) ? 1.f : 0.f;
  const int Li = (int)(Lf + 0.5f);

  float ent = 0.f;
  {
    float p = (40.f - Lf) / (40.f + 1e-8f);
    ent -= p * __logf(p + 1e-8f);
  }
  float uniq = 0.f, maxc = 0.f, minc = 1e30f, s1 = 0.f, s2 = 0.f, s3 = 0.f;
#pragma unroll 1
  for (int v = 1; v < VV; ++v) {
    float cnt = 0.f;
#pragma unroll
    for (int t = 0; t < TT; ++t) cnt += (xr[t] == v) ? 1.f : 0.f;
    float p = cnt / (40.f + 1e-8f);
    ent -= p * __logf(p + 1e-8f);
    if (cnt > 0.f) { uniq += 1.f; minc = fminf(minc, cnt); }
    maxc = fmaxf(maxc, cnt);
    if (v < 27) s1 += cnt; else if (v < 37) s2 += cnt; else s3 += cnt;
  }
  float tot = fmaxf(Lf, 1e-8f);
  float cd0 = uniq / 40.f, cd1 = maxc / tot;
  float cd2 = (minc < 1e29f) ? (minc / tot) : 0.f;
  float cd3 = s1 / tot, cd4 = s2 / tot, cd5 = s3 / tot;
  if (Li <= 0) { cd0 = cd1 = cd2 = cd3 = cd4 = cd5 = 0.f; }

  float rep = 0.f, inc = 0.f, dec = 0.f, per = 0.f;
#pragma unroll
  for (int t = 0; t < TT - 1; ++t)
    if (t < Li - 1) {
      rep += (xr[t] == xr[t + 1]) ? 1.f : 0.f;
      inc += (xr[t + 1] > xr[t]) ? 1.f : 0.f;
      dec += (xr[t + 1] < xr[t]) ? 1.f : 0.f;
    }
#pragma unroll
  for (int t = 0; t < TT - 2; ++t)
    if (t < Li - 2) per += (xr[t] == xr[t + 2]) ? 1.f : 0.f;
  float den = fmaxf(Lf - 1.f, 1.f);
  per /= fmaxf(Lf - 2.f, 1.f);
  if (Li < 4) per = 0.f;
  float p0 = rep / den, p1 = inc / den, p2 = dec / den, p3 = per;
  if (Li <= 1) { p0 = p1 = p2 = p3 = 0.f; }

  float fx[15];
  fx[0] = ent;
  fx[1] = Lf / 40.f;
  fx[2] = (Li <= 8) ? 1.f : 0.f;
  fx[3] = (Li > 8 && Li <= 16) ? 1.f : 0.f;
  fx[4] = (Li > 16) ? 1.f : 0.f;
  fx[5] = cd0; fx[6] = cd1; fx[7] = cd2; fx[8] = cd3; fx[9] = cd4; fx[10] = cd5;
  fx[11] = p0; fx[12] = p1; fx[13] = p2; fx[14] = p3;

  _Float16* row = featf16 + (size_t)b * FPAD + 128;
#pragma unroll
  for (int i = 0; i < 15; ++i) row[i] = (_Float16)fx[i];
#pragma unroll
  for (int i = 15; i < 32; ++i) row[i] = (_Float16)0.f;   // pad cols 143..159
}

// ---------------------------------------------------------------------------
// Kernel 4: MLP 143->128->64->4 + softmax, on the WMMA pipe.
//   128 WGs x 128 threads (4 waves); each wave owns one 16-row tile.
//   Layer1: 8 N-tiles x 5 K-chunks = 40 WMMAs (bias in accumulator init),
//   Layer2: 4 x 4 = 16 WMMAs; relu + D->A restage through wave-private LDS.
//   Layer3 (64->4) + softmax is tiny -> per-lane VALU.
//   Weights live in dynamic LDS in B-fragment-friendly [N][K] f16 order.
// ---------------------------------------------------------------------------
#define W1SZ (128 * FPAD)          // halves
#define W2SZ (64 * 128)            // halves
#define SM_W1 0
#define SM_W2 (SM_W1 + W1SZ * 2)                 // 40960
#define SM_W3 (SM_W2 + W2SZ * 2)                 // 57344  (f32 [4][64])
#define SM_B1 (SM_W3 + 256 * 4)                  // 58368  (f32 [128])
#define SM_B2 (SM_B1 + 128 * 4)                  // 58880  (f32 [64])
#define SM_B3 (SM_B2 + 64 * 4)                   // 59136  (f32 [4], pad 16)
#define SM_H1 (SM_B3 + 16)                       // 59152  (f16 [4 waves][16][128])
#define SM_TOTAL (SM_H1 + 4 * 16 * 128 * 2)      // 75536 bytes

__global__ __launch_bounds__(128) void mlp_kernel(const _Float16* __restrict__ featf16,
                                                  const float* __restrict__ W1, const float* __restrict__ b1,
                                                  const float* __restrict__ W2, const float* __restrict__ b2,
                                                  const float* __restrict__ W3, const float* __restrict__ b3,
                                                  float* __restrict__ out) {
  extern __shared__ __align__(16) unsigned char smem[];
  _Float16* w1h  = (_Float16*)(smem + SM_W1);
  _Float16* w2h  = (_Float16*)(smem + SM_W2);
  float*    sw3  = (float*)(smem + SM_W3);
  float*    sb1  = (float*)(smem + SM_B1);
  float*    sb2  = (float*)(smem + SM_B2);
  float*    sb3  = (float*)(smem + SM_B3);

  const int tid    = threadIdx.x;
  const int lane   = tid & 31;
  const int w      = tid >> 5;
  const int m      = lane & 15;
  const int halfId = lane >> 4;
  const int tile   = blockIdx.x * 4 + w;

  // stage weights: w1h[N][K] = W1[N*143+K] (K<143, else 0); w2h = W2 (already [64][128])
  for (int i = tid; i < W1SZ; i += 128) {
    int n = i / FPAD, k = i % FPAD;
    w1h[i] = (k < 143) ? (_Float16)W1[n * 143 + k] : (_Float16)0.f;
  }
  for (int i = tid; i < W2SZ; i += 128) w2h[i] = (_Float16)W2[i];
  for (int i = tid; i < 256; i += 128) sw3[i] = W3[i];
  if (tid < 128) sb1[tid] = b1[tid];
  if (tid < 64)  sb2[tid] = b2[tid];
  if (tid < 4)   sb3[tid] = b3[tid];
  __syncthreads();

  // ---- layer 1: feats[16x160] @ W1^T -> h1[16x128] ----
  Frag Af[5];
#pragma unroll
  for (int cc = 0; cc < 5; ++cc)
#pragma unroll
    for (int v = 0; v < 8; ++v) {
      int K = cc * 32 + a_kbase(v, halfId);
      Af[cc].u[v] = *(const unsigned*)(featf16 + (size_t)(tile * 16 + m) * FPAD + K);
    }
  v8f acc1[8];
#pragma unroll
  for (int nt = 0; nt < 8; ++nt) {
    float bv = sb1[nt * 16 + m];
    v8f a = {bv, bv, bv, bv, bv, bv, bv, bv};
#pragma unroll
    for (int kc = 0; kc < 5; ++kc) {
      Frag Bfr;
      const uint4* p = (const uint4*)(w1h + (nt * 16 + m) * FPAD + kc * 32 + halfId * 16);
      Bfr.q[0] = p[0];
      Bfr.q[1] = p[1];
      a = wmma_f16(Af[kc], Bfr, a);
    }
    acc1[nt] = a;
  }
  // relu + D->A restage (wave-private LDS; DS ops from one wave stay in order)
  _Float16* h1s = (_Float16*)(smem + SM_H1) + w * (16 * 128);
#pragma unroll
  for (int nt = 0; nt < 8; ++nt)
#pragma unroll
    for (int r = 0; r < 8; ++r)
      h1s[(r + 8 * halfId) * 128 + nt * 16 + m] = (_Float16)fmaxf(acc1[nt][r], 0.f);

  // ---- layer 2: h1[16x128] @ W2^T -> h2[16x64] ----
  Frag A2[4];
#pragma unroll
  for (int cc = 0; cc < 4; ++cc)
#pragma unroll
    for (int v = 0; v < 8; ++v) {
      int K = cc * 32 + a_kbase(v, halfId);
      A2[cc].u[v] = *(const unsigned*)&h1s[m * 128 + K];
    }
  v8f acc2[4];
#pragma unroll
  for (int nt = 0; nt < 4; ++nt) {
    float bv = sb2[nt * 16 + m];
    v8f a = {bv, bv, bv, bv, bv, bv, bv, bv};
#pragma unroll
    for (int kc = 0; kc < 4; ++kc) {
      Frag Bfr;
      const uint4* p = (const uint4*)(w2h + (nt * 16 + m) * 128 + kc * 32 + halfId * 16);
      Bfr.q[0] = p[0];
      Bfr.q[1] = p[1];
      a = wmma_f16(A2[kc], Bfr, a);
    }
    acc2[nt] = a;
  }
  // relu + stage h2 (reuse h1 region)
  _Float16* h2s = h1s;
#pragma unroll
  for (int nt = 0; nt < 4; ++nt)
#pragma unroll
    for (int r = 0; r < 8; ++r)
      h2s[(r + 8 * halfId) * 64 + nt * 16 + m] = (_Float16)fmaxf(acc2[nt][r], 0.f);

  // ---- layer 3 (64->4) + softmax: per-lane, lanes 0..15 each own one row ----
  if (lane < 16) {
    const int row = lane;
    const int b   = tile * 16 + row;
    float lg[EE] = {sb3[0], sb3[1], sb3[2], sb3[3]};
#pragma unroll 1
    for (int k = 0; k < 64; ++k) {
      float hv = (float)h2s[row * 64 + k];
#pragma unroll
      for (int e = 0; e < EE; ++e) lg[e] += hv * sw3[e * 64 + k];
    }
    float mx = fmaxf(fmaxf(lg[0], lg[1]), fmaxf(lg[2], lg[3]));
    float es = 0.f, ev[EE];
#pragma unroll
    for (int e = 0; e < EE; ++e) { ev[e] = __expf(lg[e] - mx); es += ev[e]; }
    float inv = frcp(es);
#pragma unroll
    for (int e = 0; e < EE; ++e) out[(size_t)b * EE + e] = ev[e] * inv;
  }
}

// ---------------------------------------------------------------------------
extern "C" void kernel_launch(void* const* d_in, const int* in_sizes, int n_in,
                              void* d_out, int out_size, void* d_ws, size_t ws_size,
                              hipStream_t stream) {
  (void)in_sizes; (void)n_in; (void)out_size; (void)ws_size;
  const int*   x     = (const int*)d_in[0];
  const float* emb   = (const float*)d_in[1];
  const float* Wihf  = (const float*)d_in[2];
  const float* Whhf  = (const float*)d_in[3];
  const float* bf    = (const float*)d_in[4];
  const float* Wihb  = (const float*)d_in[5];
  const float* Whhb  = (const float*)d_in[6];
  const float* bb    = (const float*)d_in[7];
  const float* W1    = (const float*)d_in[8];
  const float* b1    = (const float*)d_in[9];
  const float* W2    = (const float*)d_in[10];
  const float* b2    = (const float*)d_in[11];
  const float* W3    = (const float*)d_in[12];
  const float* b3    = (const float*)d_in[13];

  unsigned char* ws      = (unsigned char*)d_ws;
  unsigned char* bfrag   = ws;                         // 128 KB
  _Float16*      featf16 = (_Float16*)(ws + BFRAG_BYTES);  // B*160 f16 = 2.56 MB

  setup_bfrag<<<16, 256, 0, stream>>>(emb, Wihf, Whhf, bf, Wihb, Whhb, bb, bfrag);
  lstm_kernel<<<BB / 16, 128, 0, stream>>>(x, bfrag, featf16);
  stats_kernel<<<BB / 256, 256, 0, stream>>>(x, featf16);
  mlp_kernel<<<BB / 64, 128, SM_TOTAL, stream>>>(featf16, W1, b1, W2, b2, W3, b3,
                                                 (float*)d_out);
}
